// GNNStack_21741124452941
// MI455X (gfx1250) — matrix operand order repaired and checked by reading, not compiled, not run on hardware
//
#include <hip/hip_runtime.h>
#include <hip/hip_bf16.h>

typedef __bf16 bf16;
typedef __attribute__((ext_vector_type(16))) __bf16 v16bf;
typedef __attribute__((ext_vector_type(8)))  float  v8f;

#define NN 50000
#define NE 500000
#define DN 128
#define DE 64

// ---- WMMA fragment helpers -------------------------------------------------
// A (16x32 bf16, MxK): lane L<16 holds row M=L, K = {kb+0..7, kb+16..23};
//                      lane L>=16 holds row M=L-16, K = {kb+8..15, kb+24..31}.
__device__ __forceinline__ v16bf load_a_row(const float* __restrict__ row,
                                            int kbase, int lanehi, float scale) {
  const float* p = row + kbase + (lanehi ? 8 : 0);
  v16bf a;
#pragma unroll
  for (int i = 0; i < 8; ++i) a[i] = (bf16)(p[i] * scale);
#pragma unroll
  for (int i = 0; i < 8; ++i) a[8 + i] = (bf16)(p[16 + i] * scale);
  return a;
}

// B (32x16 bf16, KxN) from LDS weight panel stored [N][K] (bf16):
// lane L: column n, K = kb + (L<16 ? {0..7,16..23} : {8..15,24..31}).
__device__ __forceinline__ v16bf load_b_frag(const bf16* __restrict__ wt, int K,
                                             int n, int kbase, int lanehi) {
  const bf16* p = wt + (long)n * K + kbase + (lanehi ? 8 : 0);
  v16bf b;
#pragma unroll
  for (int i = 0; i < 8; ++i) { b[i] = p[i]; b[8 + i] = p[16 + i]; }
  return b;
}

// Stage a (K,N) row-major f32 weight into LDS as [N][K] bf16, plus bias.
__device__ __forceinline__ void stage_weights(const float* __restrict__ W, bf16* lds,
                                              int K, int N,
                                              const float* __restrict__ bias,
                                              float* bias_lds, int NB) {
  for (int idx = threadIdx.x; idx < K * N; idx += blockDim.x) {
    int k = idx / N, n = idx % N;
    lds[n * K + k] = (bf16)W[idx];
  }
  for (int i = threadIdx.x; i < NB; i += blockDim.x) bias_lds[i] = bias[i];
  __syncthreads();
}

// ---- trivial utility kernels ----------------------------------------------
__global__ void fill_zero_kernel(float* p, long n) {
  long i = (long)blockIdx.x * blockDim.x + threadIdx.x;
  if (i < n) p[i] = 0.0f;
}
__global__ void copy_kernel(float* d, const float* s, long n) {
  long i = (long)blockIdx.x * blockDim.x + threadIdx.x;
  if (i < n) d[i] = s[i];
}
__global__ void count_kernel(const int* __restrict__ dst, float* cnt, int e) {
  int i = blockIdx.x * blockDim.x + threadIdx.x;
  if (i < e) unsafeAtomicAdd(&cnt[dst[i]], 1.0f);
}

// ---- layer kernels ---------------------------------------------------------
// message: relu(concat(x[src], edge) @ Wm + bm) scattered (atomic add) into agg[dst]
__global__ void __launch_bounds__(256) msg_kernel(
    const float* __restrict__ x, const float* __restrict__ ea,
    const int* __restrict__ src, const int* __restrict__ dst,
    const float* __restrict__ Wm, const float* __restrict__ bm,
    float* __restrict__ agg, int ntiles) {
  __shared__ bf16 wlds[DN * 192];
  __shared__ float blds[DN];
  stage_weights(Wm, wlds, 192, DN, bm, blds, DN);

  int wave = threadIdx.x >> 5, lane = threadIdx.x & 31;
  int lanehi = lane >> 4, lanelo = lane & 15;
  int tile = blockIdx.x * 8 + wave;
  if (tile >= ntiles) return;
  int e0 = tile * 16;
  int myedge = e0 + lanelo;
  const float* xrow = x + (long)src[myedge] * DN;
  const float* erow = ea + (long)myedge * DE;

  v8f acc[8] = {};
#pragma unroll
  for (int kk = 0; kk < 6; ++kk) {
    v16bf a = (kk < 4) ? load_a_row(xrow, kk * 32, lanehi, 1.0f)
                       : load_a_row(erow, (kk - 4) * 32, lanehi, 1.0f);
#pragma unroll
    for (int nn = 0; nn < 8; ++nn) {
      v16bf b = load_b_frag(wlds, 192, nn * 16 + lanelo, kk * 32, lanehi);
      acc[nn] = __builtin_amdgcn_wmma_f32_16x16x32_bf16(false, a, false, b,
                                                        (short)0, acc[nn], false, false);
    }
  }
  int dn[8];
#pragma unroll
  for (int v = 0; v < 8; ++v) dn[v] = dst[e0 + v + lanehi * 8];
#pragma unroll
  for (int nn = 0; nn < 8; ++nn) {
    int col = nn * 16 + lanelo;
    float bb = blds[col];
#pragma unroll
    for (int v = 0; v < 8; ++v) {
      float val = acc[nn][v] + bb;
      val = val > 0.0f ? val : 0.0f;
      unsafeAtomicAdd(agg + (long)dn[v] * DN + col, val);
    }
  }
}

// update: x = relu(concat(agg/cnt, x) @ Wa + ba)   (in place)
__global__ void __launch_bounds__(256) update_kernel(
    float* __restrict__ x, const float* __restrict__ agg,
    const float* __restrict__ cnt,
    const float* __restrict__ Wa, const float* __restrict__ ba, int ntiles) {
  __shared__ bf16 wlds[DN * 256];
  __shared__ float blds[DN];
  stage_weights(Wa, wlds, 256, DN, ba, blds, DN);

  int wave = threadIdx.x >> 5, lane = threadIdx.x & 31;
  int lanehi = lane >> 4, lanelo = lane & 15;
  int tile = blockIdx.x * 8 + wave;
  if (tile >= ntiles) return;
  int n0 = tile * 16;
  int mynode = n0 + lanelo;
  float c = cnt[mynode];
  float rc = 1.0f / (c > 1.0f ? c : 1.0f);
  const float* arow = agg + (long)mynode * DN;
  const float* xrow = x + (long)mynode * DN;

  v8f acc[8] = {};
#pragma unroll
  for (int kk = 0; kk < 8; ++kk) {
    v16bf a = (kk < 4) ? load_a_row(arow, kk * 32, lanehi, rc)
                       : load_a_row(xrow, (kk - 4) * 32, lanehi, 1.0f);
#pragma unroll
    for (int nn = 0; nn < 8; ++nn) {
      v16bf b = load_b_frag(wlds, 256, nn * 16 + lanelo, kk * 32, lanehi);
      acc[nn] = __builtin_amdgcn_wmma_f32_16x16x32_bf16(false, a, false, b,
                                                        (short)0, acc[nn], false, false);
    }
  }
#pragma unroll
  for (int nn = 0; nn < 8; ++nn) {
    int col = nn * 16 + lanelo;
    float bb = blds[col];
#pragma unroll
    for (int v = 0; v < 8; ++v) {
      float val = acc[nn][v] + bb;
      val = val > 0.0f ? val : 0.0f;
      x[(long)(n0 + v + lanehi * 8) * DN + col] = val;
    }
  }
}

// L2 normalize each 128-wide row; one wave per node.
__global__ void __launch_bounds__(256) normalize_kernel(float* __restrict__ x, int n) {
  int wave = threadIdx.x >> 5, lane = threadIdx.x & 31;
  int node = blockIdx.x * 8 + wave;
  if (node >= n) return;
  float4 v = *(const float4*)(x + (long)node * DN + lane * 4);
  float s = v.x * v.x + v.y * v.y + v.z * v.z + v.w * v.w;
#pragma unroll
  for (int off = 16; off > 0; off >>= 1) s += __shfl_xor(s, off, 32);
  float nrm = sqrtf(s);
  float sc = 1.0f / (nrm > 1e-12f ? nrm : 1e-12f);
  v.x *= sc; v.y *= sc; v.z *= sc; v.w *= sc;
  *(float4*)(x + (long)node * DN + lane * 4) = v;
}

// edge update: ea = relu(concat(x[src], x[dst], ea) @ We + be)   (in place)
__global__ void __launch_bounds__(256) edge_kernel(
    const float* __restrict__ x, float* __restrict__ ea,
    const int* __restrict__ src, const int* __restrict__ dst,
    const float* __restrict__ We, const float* __restrict__ be, int ntiles) {
  __shared__ bf16 wlds[DE * 320];
  __shared__ float blds[DE];
  stage_weights(We, wlds, 320, DE, be, blds, DE);

  int wave = threadIdx.x >> 5, lane = threadIdx.x & 31;
  int lanehi = lane >> 4, lanelo = lane & 15;
  int tile = blockIdx.x * 8 + wave;
  if (tile >= ntiles) return;
  int e0 = tile * 16;
  int myedge = e0 + lanelo;
  const float* srow = x + (long)src[myedge] * DN;
  const float* drow = x + (long)dst[myedge] * DN;
  const float* erow = ea + (long)myedge * DE;

  v8f acc[4] = {};
#pragma unroll
  for (int kk = 0; kk < 10; ++kk) {
    v16bf a = (kk < 4)   ? load_a_row(srow, kk * 32, lanehi, 1.0f)
            : (kk < 8)   ? load_a_row(drow, (kk - 4) * 32, lanehi, 1.0f)
                         : load_a_row(erow, (kk - 8) * 32, lanehi, 1.0f);
#pragma unroll
    for (int nn = 0; nn < 4; ++nn) {
      v16bf b = load_b_frag(wlds, 320, nn * 16 + lanelo, kk * 32, lanehi);
      acc[nn] = __builtin_amdgcn_wmma_f32_16x16x32_bf16(false, a, false, b,
                                                        (short)0, acc[nn], false, false);
    }
  }
#pragma unroll
  for (int nn = 0; nn < 4; ++nn) {
    int col = nn * 16 + lanelo;
    float bb = blds[col];
#pragma unroll
    for (int v = 0; v < 8; ++v) {
      float val = acc[nn][v] + bb;
      val = val > 0.0f ? val : 0.0f;
      ea[(long)(e0 + v + lanehi * 8) * DE + col] = val;
    }
  }
}

// head1: h = relu(x @ W1 + b1), 128 -> 64
__global__ void __launch_bounds__(256) head1_kernel(
    const float* __restrict__ x, const float* __restrict__ W1,
    const float* __restrict__ b1, float* __restrict__ h, int ntiles) {
  __shared__ bf16 wlds[64 * DN];
  __shared__ float blds[64];
  stage_weights(W1, wlds, DN, 64, b1, blds, 64);

  int wave = threadIdx.x >> 5, lane = threadIdx.x & 31;
  int lanehi = lane >> 4, lanelo = lane & 15;
  int tile = blockIdx.x * 8 + wave;
  if (tile >= ntiles) return;
  int n0 = tile * 16;
  const float* xrow = x + (long)(n0 + lanelo) * DN;

  v8f acc[4] = {};
#pragma unroll
  for (int kk = 0; kk < 4; ++kk) {
    v16bf a = load_a_row(xrow, kk * 32, lanehi, 1.0f);
#pragma unroll
    for (int nn = 0; nn < 4; ++nn) {
      v16bf b = load_b_frag(wlds, DN, nn * 16 + lanelo, kk * 32, lanehi);
      acc[nn] = __builtin_amdgcn_wmma_f32_16x16x32_bf16(false, a, false, b,
                                                        (short)0, acc[nn], false, false);
    }
  }
#pragma unroll
  for (int nn = 0; nn < 4; ++nn) {
    int col = nn * 16 + lanelo;
    float bb = blds[col];
#pragma unroll
    for (int v = 0; v < 8; ++v) {
      float val = acc[nn][v] + bb;
      val = val > 0.0f ? val : 0.0f;
      h[(long)(n0 + v + lanehi * 8) * 64 + col] = val;
    }
  }
}

// head2: out = h @ W2 + b2, 64 -> 128 (no relu)
__global__ void __launch_bounds__(256) head2_kernel(
    const float* __restrict__ h, const float* __restrict__ W2,
    const float* __restrict__ b2, float* __restrict__ out, int ntiles) {
  __shared__ bf16 wlds[DN * 64];
  __shared__ float blds[DN];
  stage_weights(W2, wlds, 64, DN, b2, blds, DN);

  int wave = threadIdx.x >> 5, lane = threadIdx.x & 31;
  int lanehi = lane >> 4, lanelo = lane & 15;
  int tile = blockIdx.x * 8 + wave;
  if (tile >= ntiles) return;
  int n0 = tile * 16;
  const float* hrow = h + (long)(n0 + lanelo) * 64;

  v8f acc[8] = {};
#pragma unroll
  for (int kk = 0; kk < 2; ++kk) {
    v16bf a = load_a_row(hrow, kk * 32, lanehi, 1.0f);
#pragma unroll
    for (int nn = 0; nn < 8; ++nn) {
      v16bf b = load_b_frag(wlds, 64, nn * 16 + lanelo, kk * 32, lanehi);
      acc[nn] = __builtin_amdgcn_wmma_f32_16x16x32_bf16(false, a, false, b,
                                                        (short)0, acc[nn], false, false);
    }
  }
#pragma unroll
  for (int nn = 0; nn < 8; ++nn) {
    int col = nn * 16 + lanelo;
    float bb = blds[col];
#pragma unroll
    for (int v = 0; v < 8; ++v)
      out[(long)(n0 + v + lanehi * 8) * DN + col] = acc[nn][v] + bb;
  }
}

// ---- driver ----------------------------------------------------------------
extern "C" void kernel_launch(void* const* d_in, const int* in_sizes, int n_in,
                              void* d_out, int out_size, void* d_ws, size_t ws_size,
                              hipStream_t stream) {
  const float* x_in = (const float*)d_in[0];
  const float* ea_in = (const float*)d_in[1];
  const int*   ei    = (const int*)d_in[2];
  const int*   src   = ei;
  const int*   dstv  = ei + NE;
  const float* Wm = (const float*)d_in[3];
  const float* bm = (const float*)d_in[4];
  const float* Wa = (const float*)d_in[5];
  const float* ba = (const float*)d_in[6];
  const float* We = (const float*)d_in[7];
  const float* be = (const float*)d_in[8];
  const float* W1 = (const float*)d_in[9];
  const float* b1 = (const float*)d_in[10];
  const float* W2 = (const float*)d_in[11];
  const float* b2 = (const float*)d_in[12];
  float* out = (float*)d_out;

  float* ws  = (float*)d_ws;
  float* xb  = ws;                       // NN*128
  float* agg = xb + (long)NN * DN;       // NN*128
  float* cnt = agg + (long)NN * DN;      // NN
  float* h   = cnt + NN;                 // NN*64
  float* eb  = h + (long)NN * 64;        // NE*64

  const long NX = (long)NN * DN;
  const long NEA = (long)NE * DE;
  const int ETILES = NE / 16;            // 31250
  const int NTILES = NN / 16;            // 3125
  const int EB = (ETILES + 7) / 8;
  const int NB = (NTILES + 7) / 8;

  copy_kernel<<<(int)((NX + 255) / 256), 256, 0, stream>>>(xb, x_in, NX);
  copy_kernel<<<(int)((NEA + 255) / 256), 256, 0, stream>>>(eb, ea_in, NEA);
  fill_zero_kernel<<<(NN + 255) / 256, 256, 0, stream>>>(cnt, NN);
  count_kernel<<<(NE + 255) / 256, 256, 0, stream>>>(dstv, cnt, NE);

  for (int l = 0; l < 3; ++l) {
    fill_zero_kernel<<<(int)((NX + 255) / 256), 256, 0, stream>>>(agg, NX);
    msg_kernel<<<EB, 256, 0, stream>>>(xb, eb, src, dstv,
                                       Wm + (long)l * 192 * DN, bm + l * DN,
                                       agg, ETILES);
    update_kernel<<<NB, 256, 0, stream>>>(xb, agg, cnt,
                                          Wa + (long)l * 256 * DN, ba + l * DN,
                                          NTILES);
    normalize_kernel<<<(NN + 7) / 8, 256, 0, stream>>>(xb, NN);
    edge_kernel<<<EB, 256, 0, stream>>>(xb, eb, src, dstv,
                                        We + (long)l * 320 * DE, be + l * DE,
                                        ETILES);
  }
  head1_kernel<<<NB, 256, 0, stream>>>(xb, W1, b1, h, NTILES);
  head2_kernel<<<NB, 256, 0, stream>>>(h, W2, b2, out, NTILES);
}
